// GraphAugmentedForecastingModel_33432025432721
// MI455X (gfx1250) — matrix operand order repaired and checked by reading, not compiled, not run on hardware
//
#include <hip/hip_runtime.h>
#include <hip/hip_bf16.h>

#define B_ 8
#define N_ 2000
#define L_ 96
#define H_ 64
#define G_ 32
#define F_ 5
#define E_ 64000
#define BN_ (B_*N_)
#define NG_ 256          // 4*H gate columns
#define HP_ 72           // padded LDS h-row stride in bf16 elems (16B-aligned rows)

typedef float  v8f   __attribute__((ext_vector_type(8)));
typedef __bf16 v16bf __attribute__((ext_vector_type(16)));

static __device__ __forceinline__ unsigned short f2bf(float f) {
  unsigned u = __float_as_uint(f);
  u += 0x7fffu + ((u >> 16) & 1u);           // round-to-nearest-even
  return (unsigned short)(u >> 16);
}

// Fast transcendentals: prefer gfx1250 v_tanh_f32; otherwise v_exp_f32 + v_rcp_f32
// (avoids the ~10-op IEEE division expansion the generic path produced).
static __device__ __forceinline__ float tanh_(float x) {
#if __has_builtin(__builtin_amdgcn_tanhf)
  return __builtin_amdgcn_tanhf(x);
#else
  x = fminf(15.0f, fmaxf(-15.0f, x));
  float e = __expf(2.0f * x);
  return (e - 1.0f) * __builtin_amdgcn_rcpf(e + 1.0f);
#endif
}
static __device__ __forceinline__ float sigm(float x) {
#if __has_builtin(__builtin_amdgcn_tanhf)
  return 0.5f + 0.5f * __builtin_amdgcn_tanhf(0.5f * x);
#else
  return __builtin_amdgcn_rcpf(1.0f + __expf(-x));
#endif
}

// Build a bf16 B-operand (32x16, K-major, contiguous k order matching load_a_lds) from a
// row-major f32 weight W[256][64]:  B[k][n] = W[col+n][k].
// lanes 0-15 hold K = kb*32 + 0..15 (2 per VGPR), lanes 16-31 hold +16.
static __device__ __forceinline__ v16bf load_w_bf16(const float* W, int col, int kb, int lane) {
  int n = lane & 15, kh = lane >> 4;
  const float* src = W + (size_t)(col + n) * H_ + kb * 32 + kh * 16;
  union { unsigned short s[16]; v16bf v; } r;
#pragma unroll
  for (int j = 0; j < 16; ++j) r.s[j] = f2bf(src[j]);
  return r.v;
}

// A-operand (16x32 bf16) from LDS row-major h[16][HP_]:
// lane<16: row=lane,   K = kb*32 + {0..15}
// lane>=16: row=lane-16, K = kb*32 + {16..31}
static __device__ __forceinline__ v16bf load_a_lds(const unsigned short* h, int lane, int kb) {
  int row = lane & 15;
  int kbase = ((lane >> 4) << 4) + (kb << 5);
  union { uint4 u[2]; v16bf v; } r;
  r.u[0] = *(const uint4*)(h + row * HP_ + kbase);
  r.u[1] = *(const uint4*)(h + row * HP_ + kbase + 8);
  return r.v;
}

__global__ __launch_bounds__(256) void lstm_wmma_kernel(
    const float* __restrict__ xg,
    const float* __restrict__ Wih0, const float* __restrict__ Whh0,
    const float* __restrict__ bih0, const float* __restrict__ bhh0,
    const float* __restrict__ Wih1, const float* __restrict__ Whh1,
    const float* __restrict__ bih1, const float* __restrict__ bhh1,
    float* __restrict__ h1o)
{
  __shared__ __align__(16) float xs[16][L_];              // x tile (16 series x 96 steps)
  __shared__ __align__(16) float gbuf[16][NG_ + 1];       // gate buffer (f32)
  __shared__ __align__(16) unsigned short h0s[16 * HP_];  // layer-0 hidden (bf16)
  __shared__ __align__(16) unsigned short h1s[16 * HP_];  // layer-1 hidden (bf16)

  const int tid = threadIdx.x;
  const int lane = tid & 31;
  const int wv = tid >> 5;                 // wave 0..7
  const int cb = wv * 32;                  // this wave's gate-column base
  const int s0 = blockIdx.x * 16;          // first series handled by this block
  const int rowbase = (lane >> 4) << 3;    // C/D-layout row base for this lane
  const int nn = lane & 15;                // C/D-layout column within tile

  // stage x tile into LDS
  for (int i = tid; i < 16 * L_; i += 256) {
    int r = i / L_, t = i % L_;
    xs[r][t] = xg[(size_t)(s0 + r) * L_ + t];
  }
  // zero hidden state
  for (int i = tid; i < 16 * HP_; i += 256) { h0s[i] = 0; h1s[i] = 0; }

  // persistent bf16 weights in registers (B-operand layout), 12 x v16bf
  const int col0 = cb, col1 = cb + 16;
  v16bf wh0_00 = load_w_bf16(Whh0, col0, 0, lane);
  v16bf wh0_01 = load_w_bf16(Whh0, col0, 1, lane);
  v16bf wh0_10 = load_w_bf16(Whh0, col1, 0, lane);
  v16bf wh0_11 = load_w_bf16(Whh0, col1, 1, lane);
  v16bf wi1_00 = load_w_bf16(Wih1, col0, 0, lane);
  v16bf wi1_01 = load_w_bf16(Wih1, col0, 1, lane);
  v16bf wi1_10 = load_w_bf16(Wih1, col1, 0, lane);
  v16bf wi1_11 = load_w_bf16(Wih1, col1, 1, lane);
  v16bf wh1_00 = load_w_bf16(Whh1, col0, 0, lane);
  v16bf wh1_01 = load_w_bf16(Whh1, col0, 1, lane);
  v16bf wh1_10 = load_w_bf16(Whh1, col1, 0, lane);
  v16bf wh1_11 = load_w_bf16(Whh1, col1, 1, lane);

  // per-lane column params (x-coupling weight + fused biases)
  const int c0g = col0 + nn, c1g = col1 + nn;
  const float wx0 = Wih0[c0g], wx1 = Wih0[c1g];
  const float b0a = bih0[c0g] + bhh0[c0g];
  const float b0b = bih0[c1g] + bhh0[c1g];
  const float b1a = bih1[c0g] + bhh1[c0g];
  const float b1b = bih1[c1g] + bhh1[c1g];

  float c0r[4] = {0.f, 0.f, 0.f, 0.f};     // cell states live in registers
  float c1r[4] = {0.f, 0.f, 0.f, 0.f};

  for (int t = 0; t < L_; ++t) {
    __syncthreads();
    // ---------------- layer 0: gates = x*Wih0 + h0 @ Whh0^T + b ----------------
    {
      v16bf a0 = load_a_lds(h0s, lane, 0);
      v16bf a1 = load_a_lds(h0s, lane, 1);
      v8f acc0, acc1;
#pragma unroll
      for (int v = 0; v < 8; ++v) {
        float xv = xs[rowbase + v][t];
        acc0[v] = xv * wx0 + b0a;
        acc1[v] = xv * wx1 + b0b;
      }
      acc0 = __builtin_amdgcn_wmma_f32_16x16x32_bf16(false, a0, false, wh0_00, (short)0, acc0, false, false);
      acc0 = __builtin_amdgcn_wmma_f32_16x16x32_bf16(false, a1, false, wh0_01, (short)0, acc0, false, false);
      acc1 = __builtin_amdgcn_wmma_f32_16x16x32_bf16(false, a0, false, wh0_10, (short)0, acc1, false, false);
      acc1 = __builtin_amdgcn_wmma_f32_16x16x32_bf16(false, a1, false, wh0_11, (short)0, acc1, false, false);
#pragma unroll
      for (int v = 0; v < 8; ++v) {
        gbuf[rowbase + v][c0g] = acc0[v];
        gbuf[rowbase + v][c1g] = acc1[v];
      }
    }
    __syncthreads();
    // ---------------- layer 0 elementwise (i,f,g,o -> c0,h0) ----------------
#pragma unroll
    for (int k = 0; k < 4; ++k) {
      int i = tid + k * 256;
      int r = i >> 6, j = i & 63;
      float ig = gbuf[r][j], fg = gbuf[r][64 + j], gg = gbuf[r][128 + j], og = gbuf[r][192 + j];
      float c = sigm(fg) * c0r[k] + sigm(ig) * tanh_(gg);
      c0r[k] = c;
      float h = sigm(og) * tanh_(c);
      h0s[r * HP_ + j] = f2bf(h);
    }
    __syncthreads();
    // ---------------- layer 1: gates = h0 @ Wih1^T + h1 @ Whh1^T + b ----------------
    {
      v16bf a0 = load_a_lds(h0s, lane, 0);
      v16bf a1 = load_a_lds(h0s, lane, 1);
      v16bf a2 = load_a_lds(h1s, lane, 0);
      v16bf a3 = load_a_lds(h1s, lane, 1);
      v8f acc0, acc1;
#pragma unroll
      for (int v = 0; v < 8; ++v) { acc0[v] = b1a; acc1[v] = b1b; }
      acc0 = __builtin_amdgcn_wmma_f32_16x16x32_bf16(false, a0, false, wi1_00, (short)0, acc0, false, false);
      acc0 = __builtin_amdgcn_wmma_f32_16x16x32_bf16(false, a1, false, wi1_01, (short)0, acc0, false, false);
      acc0 = __builtin_amdgcn_wmma_f32_16x16x32_bf16(false, a2, false, wh1_00, (short)0, acc0, false, false);
      acc0 = __builtin_amdgcn_wmma_f32_16x16x32_bf16(false, a3, false, wh1_01, (short)0, acc0, false, false);
      acc1 = __builtin_amdgcn_wmma_f32_16x16x32_bf16(false, a0, false, wi1_10, (short)0, acc1, false, false);
      acc1 = __builtin_amdgcn_wmma_f32_16x16x32_bf16(false, a1, false, wi1_11, (short)0, acc1, false, false);
      acc1 = __builtin_amdgcn_wmma_f32_16x16x32_bf16(false, a2, false, wh1_10, (short)0, acc1, false, false);
      acc1 = __builtin_amdgcn_wmma_f32_16x16x32_bf16(false, a3, false, wh1_11, (short)0, acc1, false, false);
#pragma unroll
      for (int v = 0; v < 8; ++v) {
        gbuf[rowbase + v][c0g] = acc0[v];
        gbuf[rowbase + v][c1g] = acc1[v];
      }
    }
    __syncthreads();
    // ---------------- layer 1 elementwise (-> c1,h1) ----------------
#pragma unroll
    for (int k = 0; k < 4; ++k) {
      int i = tid + k * 256;
      int r = i >> 6, j = i & 63;
      float ig = gbuf[r][j], fg = gbuf[r][64 + j], gg = gbuf[r][128 + j], og = gbuf[r][192 + j];
      float c = sigm(fg) * c1r[k] + sigm(ig) * tanh_(gg);
      c1r[k] = c;
      float h = sigm(og) * tanh_(c);
      h1s[r * HP_ + j] = f2bf(h);
      if (t == L_ - 1) h1o[(size_t)(s0 + r) * H_ + j] = h;  // h_n of last layer
    }
  }
}

// ======================= GCN kernels (bandwidth-trivial) =======================
__global__ void deg_init_kernel(float* deg) {
  int n = blockIdx.x * blockDim.x + threadIdx.x;
  if (n < N_) deg[n] = 1.0f;  // self-loop
}
__global__ void deg_edge_kernel(const int* __restrict__ ei, float* deg) {
  int e = blockIdx.x * blockDim.x + threadIdx.x;
  if (e < E_) atomicAdd(&deg[ei[E_ + e]], 1.0f);
}
__global__ void dinv_kernel(const float* __restrict__ deg, float* dinv) {
  int n = blockIdx.x * blockDim.x + threadIdx.x;
  if (n < N_) dinv[n] = rsqrtf(deg[n]);
}
__global__ void xw_first_kernel(const float* __restrict__ nf, const float* __restrict__ Wg1, float* xw) {
  int i = blockIdx.x * blockDim.x + threadIdx.x;
  if (i < N_ * G_) { int n = i >> 5, k = i & 31; xw[i] = nf[n] * Wg1[k]; }
}
__global__ void xw_dense_kernel(const float* __restrict__ gin, const float* __restrict__ W, float* xw) {
  int i = blockIdx.x * blockDim.x + threadIdx.x;
  if (i < N_ * G_) {
    int n = i >> 5, k = i & 31;
    float acc = 0.f;
#pragma unroll
    for (int j = 0; j < G_; ++j) acc += gin[n * G_ + j] * W[j * G_ + k];
    xw[i] = acc;
  }
}
__global__ void gcn_self_kernel(const float* __restrict__ xw, const float* __restrict__ dinv, float* out) {
  int i = blockIdx.x * blockDim.x + threadIdx.x;
  if (i < N_ * G_) { int n = i >> 5; float d = dinv[n]; out[i] = xw[i] * d * d; }
}
__global__ void gcn_scatter_kernel(const int* __restrict__ ei, const float* __restrict__ xw,
                                   const float* __restrict__ dinv, float* out) {
  int i = blockIdx.x * blockDim.x + threadIdx.x;
  if (i < E_ * G_) {
    int e = i >> 5, k = i & 31;
    int s = ei[e], d = ei[E_ + e];
    atomicAdd(&out[d * G_ + k], xw[s * G_ + k] * (dinv[s] * dinv[d]));
  }
}
__global__ void gcn_bias_relu_kernel(float* g, const float* __restrict__ b) {
  int i = blockIdx.x * blockDim.x + threadIdx.x;
  if (i < N_ * G_) { int k = i & 31; g[i] = fmaxf(g[i] + b[k], 0.0f); }
}

// ======================= final FC: (H+G=96) -> F=5 =======================
__global__ void fc_kernel(const float* __restrict__ h1o, const float* __restrict__ g,
                          const float* __restrict__ Wfc, const float* __restrict__ bfc,
                          float* __restrict__ out) {
  int s = blockIdx.x * blockDim.x + threadIdx.x;
  if (s < BN_) {
    int n = s % N_;
    const float* hr = h1o + (size_t)s * H_;
    const float* gr = g + (size_t)n * G_;
#pragma unroll
    for (int f = 0; f < F_; ++f) {
      float acc = bfc[f];
      const float* wf = Wfc + f * (H_ + G_);
#pragma unroll
      for (int j = 0; j < H_; ++j) acc += hr[j] * wf[j];
#pragma unroll
      for (int k = 0; k < G_; ++k) acc += gr[k] * wf[H_ + k];
      out[(size_t)s * F_ + f] = acc;
    }
  }
}

extern "C" void kernel_launch(void* const* d_in, const int* in_sizes, int n_in,
                              void* d_out, int out_size, void* d_ws, size_t ws_size,
                              hipStream_t stream) {
  const float* xg   = (const float*)d_in[0];
  const float* nf   = (const float*)d_in[1];
  const int*   ei   = (const int*)d_in[2];
  const float* Wih0 = (const float*)d_in[3];
  const float* Whh0 = (const float*)d_in[4];
  const float* bih0 = (const float*)d_in[5];
  const float* bhh0 = (const float*)d_in[6];
  const float* Wih1 = (const float*)d_in[7];
  const float* Whh1 = (const float*)d_in[8];
  const float* bih1 = (const float*)d_in[9];
  const float* bhh1 = (const float*)d_in[10];
  const float* Wg1  = (const float*)d_in[11];
  const float* bg1  = (const float*)d_in[12];
  const float* Wg2  = (const float*)d_in[13];
  const float* bg2  = (const float*)d_in[14];
  const float* Wg3  = (const float*)d_in[15];
  const float* bg3  = (const float*)d_in[16];
  const float* Wfc  = (const float*)d_in[17];
  const float* bfc  = (const float*)d_in[18];
  float* out = (float*)d_out;

  // workspace layout (floats)
  float* ws   = (float*)d_ws;
  float* h1o  = ws;                        // BN*H = 1,024,000
  float* deg  = h1o + (size_t)BN_ * H_;    // N (padded slot 2048)
  float* dinv = deg + 2048;                // N (padded slot 2048)
  float* ga   = dinv + 2048;               // N*G
  float* gb   = ga + N_ * G_;              // N*G
  float* xw   = gb + N_ * G_;              // N*G

  dim3 blk(256);
  lstm_wmma_kernel<<<BN_ / 16, blk, 0, stream>>>(xg, Wih0, Whh0, bih0, bhh0,
                                                 Wih1, Whh1, bih1, bhh1, h1o);

  int gN  = (N_ + 255) / 256;
  int gE  = (E_ + 255) / 256;
  int gNG = (N_ * G_ + 255) / 256;
  int gEG = (E_ * G_ + 255) / 256;

  deg_init_kernel<<<gN, blk, 0, stream>>>(deg);
  deg_edge_kernel<<<gE, blk, 0, stream>>>(ei, deg);
  dinv_kernel<<<gN, blk, 0, stream>>>(deg, dinv);

  // GCN layer 1
  xw_first_kernel<<<gNG, blk, 0, stream>>>(nf, Wg1, xw);
  gcn_self_kernel<<<gNG, blk, 0, stream>>>(xw, dinv, ga);
  gcn_scatter_kernel<<<gEG, blk, 0, stream>>>(ei, xw, dinv, ga);
  gcn_bias_relu_kernel<<<gNG, blk, 0, stream>>>(ga, bg1);
  // GCN layer 2
  xw_dense_kernel<<<gNG, blk, 0, stream>>>(ga, Wg2, xw);
  gcn_self_kernel<<<gNG, blk, 0, stream>>>(xw, dinv, gb);
  gcn_scatter_kernel<<<gEG, blk, 0, stream>>>(ei, xw, dinv, gb);
  gcn_bias_relu_kernel<<<gNG, blk, 0, stream>>>(gb, bg2);
  // GCN layer 3
  xw_dense_kernel<<<gNG, blk, 0, stream>>>(gb, Wg3, xw);
  gcn_self_kernel<<<gNG, blk, 0, stream>>>(xw, dinv, ga);
  gcn_scatter_kernel<<<gEG, blk, 0, stream>>>(ei, xw, dinv, ga);
  gcn_bias_relu_kernel<<<gNG, blk, 0, stream>>>(ga, bg3);

  fc_kernel<<<(BN_ + 255) / 256, blk, 0, stream>>>(h1o, ga, Wfc, bfc, out);
}
